// _EdgeUpdateMLP_63488206569665
// MI455X (gfx1250) — compile-verified, matrix operand order checked by reading, not run on hardware
//
#include <hip/hip_runtime.h>
#include <hip/hip_bf16.h>

// ---------- types for CDNA5 WMMA ----------
typedef __attribute__((ext_vector_type(16))) __bf16 v16bf;
typedef __attribute__((ext_vector_type(8)))  __bf16 bf16x8;
typedef __attribute__((ext_vector_type(8)))  float  v8f;

#define MT       128   // edges per block (M tile)
#define IN_DIM   384
#define OUT_DIM  128
#define LDA      392   // lA row stride (bf16 elems), 784B = 16B aligned, bank-skewed
#define LDH      136   // lH row stride (bf16 elems), 272B = 16B aligned
#define LN_EPS   1e-5f

// Swizzled-weight workspace layout (dwords = packed bf16 pairs):
//   W1: 8 n-tiles x 12 k-tiles x 32 lanes x 8 dwords = 24576 dwords
//   W2: 8 n-tiles x  4 k-tiles x 32 lanes x 8 dwords =  8192 dwords
#define W1_FRAG_DW  24576
#define W2_FRAG_DW  8192
#define WS_DWORDS   (W1_FRAG_DW + W2_FRAG_DW)

__device__ __forceinline__ unsigned short f32_to_bf16(float f) {
  unsigned u = __builtin_bit_cast(unsigned, f);
  unsigned r = u + 0x7FFFu + ((u >> 16) & 1u);   // round-to-nearest-even
  return (unsigned short)(r >> 16);
}

// packed f32x2 -> bf16x2 as one dword; uses v_cvt_pk_bf16_f32 when available
__device__ __forceinline__ unsigned pack_bf16x2(float a, float b) {
#if __has_builtin(__builtin_amdgcn_cvt_pk_bf16_f32)
  typedef __attribute__((ext_vector_type(2))) __bf16 v2bf;
  v2bf r = __builtin_amdgcn_cvt_pk_bf16_f32(a, b);
  return __builtin_bit_cast(unsigned, r);
#else
  return (unsigned)f32_to_bf16(a) | ((unsigned)f32_to_bf16(b) << 16);
#endif
}

__device__ __forceinline__ unsigned short bf16_lo(float a) {
#if __has_builtin(__builtin_amdgcn_cvt_pk_bf16_f32)
  return (unsigned short)(pack_bf16x2(a, a) & 0xFFFFu);
#else
  return f32_to_bf16(a);
#endif
}

union AFrag { v16bf v; bf16x8 h[2]; };
union BFrag { v16bf v; uint4 q[2]; unsigned short u[16]; };

// A-fragment (16x32 bf16) from an LDS row-major bf16 tile.
// ISA 7.12.2: lanes 0-15 = rows, VGPR0-3 K=0..7 (contig), VGPR4-7 K=16..23;
// lanes 16-31 same rows, K ranges +8.  -> two 16-byte LDS loads per lane.
__device__ __forceinline__ v16bf load_a_frag(const unsigned short* base, int ld,
                                             int mt, int kbase, int lane) {
  int m = lane & 15, half = lane >> 4;
  const unsigned short* r = base + (mt * 16 + m) * ld + kbase + half * 8;
  AFrag f;
  f.h[0] = *(const bf16x8*)(r);
  f.h[1] = *(const bf16x8*)(r + 16);
  return f.v;
}

// B-fragment from pre-swizzled bf16 workspace: two coalesced b128 loads.
__device__ __forceinline__ v16bf load_b_swz(const unsigned* __restrict__ ws,
                                            int fragIdx, int lane) {
  const uint4* p = (const uint4*)(ws + (fragIdx * 32 + lane) * 8);
  BFrag f;
  f.q[0] = p[0];
  f.q[1] = p[1];
  return f.v;
}

// Fallback: B-fragment built from row-major f32 weights [K][128].
__device__ __forceinline__ v16bf load_b_direct(const float* __restrict__ W,
                                               int ktBase, int nBase, int lane) {
  int n = nBase + (lane & 15);
  int half = lane >> 4;
  BFrag f;
#pragma unroll
  for (int g = 0; g < 8; ++g) {
    int k = ktBase + ((g >= 4) ? 16 : 0) + half * 8 + (g & 3) * 2;
    f.u[2 * g]     = bf16_lo(W[k * OUT_DIM + n]);
    f.u[2 * g + 1] = bf16_lo(W[(k + 1) * OUT_DIM + n]);
  }
  return f.v;
}

// ---- prep kernel: swizzle W1/W2 (f32) into bf16 fragment layout in ws ----
__global__ void __launch_bounds__(256)
prep_weights_kernel(const float* __restrict__ W1, const float* __restrict__ W2,
                    unsigned* __restrict__ ws) {
  int i = blockIdx.x * 256 + threadIdx.x;       // 0 .. WS_DWORDS-1
  const float* W;
  int kt, nt, r;
  if (i < W1_FRAG_DW) {
    W = W1; r = i;
    int frag = r >> 8;                          // /(32*8)
    kt = frag % 12; nt = frag / 12;
  } else {
    W = W2; r = i - W1_FRAG_DW;
    int frag = r >> 8;
    kt = frag & 3; nt = frag >> 2;
  }
  int g    = r & 7;
  int lane = (r >> 3) & 31;
  int n    = nt * 16 + (lane & 15);
  int half = lane >> 4;
  int k    = kt * 32 + ((g >= 4) ? 16 : 0) + half * 8 + (g & 3) * 2;
  ws[i] = pack_bf16x2(W[k * OUT_DIM + n], W[(k + 1) * OUT_DIM + n]);
}

__global__ void __launch_bounds__(256)
edge_mlp_kernel(const float* __restrict__ node,
                const int*   __restrict__ eidx,
                const float* __restrict__ edgef,
                const float* __restrict__ W1, const float* __restrict__ b1,
                const float* __restrict__ W2, const float* __restrict__ b2,
                const float* __restrict__ gamma_, const float* __restrict__ beta_,
                float* __restrict__ out,
                const unsigned* __restrict__ wsw, int useSwz, int E) {
  __shared__ __align__(16) unsigned short lA[MT * LDA];  // bf16 A-tile; reused as f32 H2
  __shared__ __align__(16) unsigned short lH[MT * LDH];  // bf16 intermediate
  __shared__ float lMu[MT];
  __shared__ float lRs[MT];

  const int tid  = threadIdx.x;
  const int lane = tid & 31;
  const int wv   = tid >> 5;          // 0..7, owns N columns [16*wv, 16*wv+16)
  const int nBase = wv * 16;
  const int half = lane >> 4;
  const long long e0 = (long long)blockIdx.x * MT;

  const int* srcI = eidx;
  const int* dstI = eidx + E;

  // ---- first two W1 fragments issued now: latency hides under the gather --
  v16bf bC, bN, bF;
  if (useSwz) {
    bC = load_b_swz(wsw, wv * 12 + 0, lane);
    bN = load_b_swz(wsw, wv * 12 + 1, lane);
  } else {
    bC = load_b_direct(W1, 0, nBase, lane);
    bN = load_b_direct(W1, 32, nBase, lane);
  }
  bF = bN;

  // ---------------- Stage 1: gather + concat + f32->bf16 into LDS ----------
  // 128 rows x 96 float4 chunks = 12288 chunks / 256 threads = 48 each
#pragma unroll 4
  for (int j = 0; j < 48; ++j) {
    int idx = tid + 256 * j;
    int row = idx / 96;
    int c4  = idx - row * 96;          // 0..95 (float4 index within 384)
    long long e = e0 + row;
    float4 v = make_float4(0.f, 0.f, 0.f, 0.f);
    if (e < E) {
      int seg  = c4 >> 5;              // 0: src, 1: dst, 2: edge
      int coff = (c4 & 31) << 2;
      const float* p;
      if (seg == 0)      p = node + (long long)srcI[e] * 128 + coff;
      else if (seg == 1) p = node + (long long)dstI[e] * 128 + coff;
      else               p = edgef + e * 128 + coff;
      v = *(const float4*)p;
    }
    *(uint2*)&lA[row * LDA + c4 * 4] =
        make_uint2(pack_bf16x2(v.x, v.y), pack_bf16x2(v.z, v.w));
  }
  __syncthreads();

  // ---------------- Stage 2: GEMM1  [128x384] x [384x16-strip] ------------
  const v8f vzero = {0.f, 0.f, 0.f, 0.f, 0.f, 0.f, 0.f, 0.f};
  v8f acc[8];
#pragma unroll
  for (int mt = 0; mt < 8; ++mt) acc[mt] = vzero;

  // kt-outer / mt-inner, 2-deep A-fragment pipeline + 3-slot streaming B
  // prefetch (kt+2 issued at each group start). Low register pressure keeps
  // everything in the direct 256-VGPR window (no s_set_vgpr_msb churn).
  {
    v16bf a0 = load_a_frag(lA, LDA, 0, 0, lane);
    v16bf a1 = load_a_frag(lA, LDA, 1, 0, lane);
#pragma unroll
    for (int it = 0; it < 96; ++it) {
      int kt = it >> 3, mt = it & 7;
      if ((it & 7) == 0) {
        int kf = (kt + 2 <= 11) ? kt + 2 : 11;
        bF = useSwz ? load_b_swz(wsw, wv * 12 + kf, lane)
                    : load_b_direct(W1, kf * 32, nBase, lane);
      }
      int ip = (it + 2 < 96) ? it + 2 : it;     // clamped prefetch index
      v16bf an = load_a_frag(lA, LDA, ip & 7, (ip >> 3) * 32, lane);
      acc[mt] = __builtin_amdgcn_wmma_f32_16x16x32_bf16(
          false, a0, false, bC, (short)0, acc[mt], false, false);
      a0 = a1; a1 = an;
      if ((it & 7) == 7) { bC = bN; bN = bF; }
    }
  }

  // B2 fragments issued before the epilogue/barrier: latency overlaps them
  v16bf B2[4];
  if (useSwz) {
#pragma unroll
    for (int kt = 0; kt < 4; ++kt)
      B2[kt] = load_b_swz(wsw + W1_FRAG_DW, wv * 4 + kt, lane);
  } else {
#pragma unroll
    for (int kt = 0; kt < 4; ++kt) B2[kt] = load_b_direct(W2, kt * 32, nBase, lane);
  }

  // bias + ReLU + bf16 store of H strip to LDS
  {
    float bias1 = b1[nBase + (lane & 15)];
#pragma unroll
    for (int mt = 0; mt < 8; ++mt) {
#pragma unroll
      for (int g = 0; g < 8; ++g) {
        float x = acc[mt][g] + bias1;
        x = x > 0.f ? x : 0.f;
        int row = mt * 16 + g + 8 * half;
        lH[row * LDH + nBase + (lane & 15)] = bf16_lo(x);
      }
    }
  }
  __syncthreads();

  // ---------------- Stage 3: GEMM2  [128x128] x [128x16-strip] ------------
  v8f acc2[8];
#pragma unroll
  for (int mt = 0; mt < 8; ++mt) acc2[mt] = vzero;

  {
    v16bf a0 = load_a_frag(lH, LDH, 0, 0, lane);
    v16bf a1 = load_a_frag(lH, LDH, 1, 0, lane);
#pragma unroll
    for (int it = 0; it < 32; ++it) {
      int kt = it >> 3, mt = it & 7;
      int ip = (it + 2 < 32) ? it + 2 : it;
      v16bf an = load_a_frag(lH, LDH, ip & 7, (ip >> 3) * 32, lane);
      acc2[mt] = __builtin_amdgcn_wmma_f32_16x16x32_bf16(
          false, a0, false, B2[kt], (short)0, acc2[mt], false, false);
      a0 = a1; a1 = an;
    }
  }

  // bias2 + f32 store to LDS (reuse lA space: 128*128*4 = 64KB <= 100KB)
  float* H2 = (float*)lA;
  {
    float bias2 = b2[nBase + (lane & 15)];
#pragma unroll
    for (int mt = 0; mt < 8; ++mt) {
#pragma unroll
      for (int g = 0; g < 8; ++g) {
        int row = mt * 16 + g + 8 * half;
        H2[row * OUT_DIM + nBase + (lane & 15)] = acc2[mt][g] + bias2;
      }
    }
  }
  __syncthreads();

  // ---------------- Stage 4: LayerNorm stats (one thread per row) ---------
  if (tid < MT) {
    const float4* r4 = (const float4*)(H2 + tid * OUT_DIM);
    float s = 0.f, ss = 0.f;
#pragma unroll 8
    for (int c = 0; c < OUT_DIM / 4; ++c) {
      float4 x = r4[c];
      s  += x.x + x.y + x.z + x.w;
      ss += x.x * x.x + x.y * x.y + x.z * x.z + x.w * x.w;
    }
    float mu  = s * (1.f / OUT_DIM);
    float var = ss * (1.f / OUT_DIM) - mu * mu;
    lMu[tid] = mu;
    lRs[tid] = rsqrtf(var + LN_EPS);
  }
  __syncthreads();

  // ---------------- Stage 5: normalized, coalesced float4 store -----------
#pragma unroll 4
  for (int j = 0; j < 16; ++j) {
    int idx = tid + 256 * j;          // 0..4095
    int row = idx >> 5;
    int c   = (idx & 31) << 2;        // float4 column
    long long e = e0 + row;
    if (e < E) {
      float mu = lMu[row], rs = lRs[row];
      float4 x  = *(const float4*)(H2 + row * OUT_DIM + c);
      float4 g4 = *(const float4*)(gamma_ + c);
      float4 b4 = *(const float4*)(beta_ + c);
      float4 o;
      o.x = (x.x - mu) * rs * g4.x + b4.x;
      o.y = (x.y - mu) * rs * g4.y + b4.y;
      o.z = (x.z - mu) * rs * g4.z + b4.z;
      o.w = (x.w - mu) * rs * g4.w + b4.w;
      *(float4*)(out + e * OUT_DIM + c) = o;
    }
  }
}

extern "C" void kernel_launch(void* const* d_in, const int* in_sizes, int n_in,
                              void* d_out, int out_size, void* d_ws, size_t ws_size,
                              hipStream_t stream) {
  const float* node  = (const float*)d_in[0];
  const int*   eidx  = (const int*)  d_in[1];
  const float* edgef = (const float*)d_in[2];
  const float* W1    = (const float*)d_in[3];
  const float* b1    = (const float*)d_in[4];
  const float* W2    = (const float*)d_in[5];
  const float* b2    = (const float*)d_in[6];
  const float* gam   = (const float*)d_in[7];
  const float* bet   = (const float*)d_in[8];
  float* out = (float*)d_out;

  int E = in_sizes[1] / 2;             // edge_index is [2, E]
  int blocks = (E + MT - 1) / MT;

  int useSwz = (ws_size >= (size_t)WS_DWORDS * sizeof(unsigned)) ? 1 : 0;
  if (useSwz) {
    prep_weights_kernel<<<WS_DWORDS / 256, 256, 0, stream>>>(
        W1, W2, (unsigned*)d_ws);
  }
  edge_mlp_kernel<<<blocks, 256, 0, stream>>>(node, eidx, edgef, W1, b1, W2,
                                              b2, gam, bet, out,
                                              (const unsigned*)d_ws, useSwz, E);
}
